// LlamaAttention_38817914422113
// MI455X (gfx1250) — compile-verified, matrix operand order checked by reading, not run on hardware
//
#include <hip/hip_runtime.h>
#include <hip/hip_bf16.h>

// ---------------------------------------------------------------------------
// Types
// ---------------------------------------------------------------------------
typedef __bf16 bf16;
typedef bf16  v4bf  __attribute__((ext_vector_type(4)));
typedef bf16  v8bf  __attribute__((ext_vector_type(8)));
typedef bf16  v16bf __attribute__((ext_vector_type(16)));
typedef float v4f   __attribute__((ext_vector_type(4)));
typedef float v8f   __attribute__((ext_vector_type(8)));

__device__ __forceinline__ v16bf cat16(v8bf lo, v8bf hi) {
    return __builtin_shufflevector(lo, hi, 0,1,2,3,4,5,6,7,8,9,10,11,12,13,14,15);
}

// ---------------------------------------------------------------------------
// CDNA5 async global->LDS copy (ASYNCcnt-tracked, bypasses VGPRs).
// Builtin signature (from hipcc diagnostic): param0 = int4 addrspace(1)*,
// param1 = int4 addrspace(3)*, then imm offset, imm cpol.
// Falls back to a register-path copy if the builtin is absent.
// ---------------------------------------------------------------------------
#define ASG __attribute__((address_space(1)))
#define ASL __attribute__((address_space(3)))
typedef int v4i_vs __attribute__((__vector_size__(4 * sizeof(int))));

#if defined(__AMDGCN__) && __has_builtin(__builtin_amdgcn_global_load_async_to_lds_b128)
#define HAVE_ASYNC_LDS 1
#else
#define HAVE_ASYNC_LDS 0
#endif

__device__ __forceinline__ void copy16_g2l(const bf16* g, bf16* l) {
#if HAVE_ASYNC_LDS
    __builtin_amdgcn_global_load_async_to_lds_b128(
        (ASG v4i_vs*)(g), (ASL v4i_vs*)(l), 0, 0);
#else
    *(v8bf*)l = *(const v8bf*)g;
#endif
}

__device__ __forceinline__ void async_wait0() {
#if HAVE_ASYNC_LDS
#if __has_builtin(__builtin_amdgcn_s_wait_asynccnt)
    __builtin_amdgcn_s_wait_asynccnt(0);
#else
    asm volatile("s_wait_asynccnt 0" ::: "memory");
#endif
#endif
}

// ---------------------------------------------------------------------------
// f32 -> bf16 conversion (vectorized x4)
// ---------------------------------------------------------------------------
__global__ __launch_bounds__(256) void cvt_f32_to_bf16_x4(
    const v4f* __restrict__ in, v4bf* __restrict__ out, int n4)
{
    int i = blockIdx.x * blockDim.x + threadIdx.x;
    if (i < n4) {
        v4f f = in[i];
        v4bf b;
        b[0] = (bf16)f[0]; b[1] = (bf16)f[1];
        b[2] = (bf16)f[2]; b[3] = (bf16)f[3];
        out[i] = b;
    }
}

// ---------------------------------------------------------------------------
// GEMM: Y[M,N] = A[M,K] @ W[N,K]^T   (bf16 in, f32 accumulate, OutT out)
// Block: 256 threads = 8 waves, tile 128x128, K-step 32, double-buffered LDS
// with async global->LDS copies overlapping the WMMA stream.
// Wave w computes a 32(M) x 64(N) subtile = 2x4 WMMA fragments.
// ---------------------------------------------------------------------------
#define GL_STRIDE 40   // 32 + 8 pad (bf16 units); 80B row stride, 16B aligned

template <typename OutT>
__global__ __launch_bounds__(256) void gemm_xwT(
    const bf16* __restrict__ A,   // [M,K]
    const bf16* __restrict__ W,   // [N,K]
    OutT* __restrict__ Y,         // [M,N]
    int M, int N, int K)
{
    __shared__ bf16 As[2][128 * GL_STRIDE];
    __shared__ bf16 Bs[2][128 * GL_STRIDE];

    const int tid  = threadIdx.x;
    const int lane = tid & 31;
    const int wave = tid >> 5;
    const int l16  = lane & 15;
    const int klo  = (lane < 16) ? 0 : 8;

    const int m0 = blockIdx.y * 128;
    const int n0 = blockIdx.x * 128;
    const int wm = (wave >> 1) * 32;  // 0,32,64,96
    const int wn = (wave & 1) * 64;   // 0,64

    // Per-thread staging chunks: 2 for A, 2 for B (512 x 16B chunks per tile)
    const int row_c0  = tid >> 2;            // chunk tid
    const int part_c0 = (tid & 3) * 8;
    const int row_c1  = (tid + 256) >> 2;    // chunk tid+256
    const int part_c1 = ((tid + 256) & 3) * 8;

    v8f acc[2][4];
    #pragma unroll
    for (int i = 0; i < 2; ++i)
        #pragma unroll
        for (int j = 0; j < 4; ++j)
            #pragma unroll
            for (int r = 0; r < 8; ++r) acc[i][j][r] = 0.0f;

    const int NK = K >> 5;

    // Prologue: stage tile 0 into buffer 0
    {
        copy16_g2l(A + (size_t)(m0 + row_c0) * K + part_c0, &As[0][row_c0 * GL_STRIDE + part_c0]);
        copy16_g2l(A + (size_t)(m0 + row_c1) * K + part_c1, &As[0][row_c1 * GL_STRIDE + part_c1]);
        copy16_g2l(W + (size_t)(n0 + row_c0) * K + part_c0, &Bs[0][row_c0 * GL_STRIDE + part_c0]);
        copy16_g2l(W + (size_t)(n0 + row_c1) * K + part_c1, &Bs[0][row_c1 * GL_STRIDE + part_c1]);
    }

    for (int kt = 0; kt < NK; ++kt) {
        const int buf = kt & 1;
        async_wait0();        // this wave's copies for tile kt are in LDS
        __syncthreads();      // all waves' copies visible; prev reads of buf^1 done

        // Kick off next tile into the other buffer (overlaps WMMAs below)
        if (kt + 1 < NK) {
            const int k1 = (kt + 1) << 5;
            copy16_g2l(A + (size_t)(m0 + row_c0) * K + k1 + part_c0, &As[buf ^ 1][row_c0 * GL_STRIDE + part_c0]);
            copy16_g2l(A + (size_t)(m0 + row_c1) * K + k1 + part_c1, &As[buf ^ 1][row_c1 * GL_STRIDE + part_c1]);
            copy16_g2l(W + (size_t)(n0 + row_c0) * K + k1 + part_c0, &Bs[buf ^ 1][row_c0 * GL_STRIDE + part_c0]);
            copy16_g2l(W + (size_t)(n0 + row_c1) * K + k1 + part_c1, &Bs[buf ^ 1][row_c1 * GL_STRIDE + part_c1]);
        }

        v16bf afrag[2], bfrag[4];
        #pragma unroll
        for (int mt = 0; mt < 2; ++mt) {
            int row = wm + mt * 16 + l16;
            v8bf lo = *(const v8bf*)(&As[buf][row * GL_STRIDE + klo]);
            v8bf hi = *(const v8bf*)(&As[buf][row * GL_STRIDE + 16 + klo]);
            afrag[mt] = cat16(lo, hi);
        }
        #pragma unroll
        for (int nt = 0; nt < 4; ++nt) {
            int row = wn + nt * 16 + l16;
            v8bf lo = *(const v8bf*)(&Bs[buf][row * GL_STRIDE + klo]);
            v8bf hi = *(const v8bf*)(&Bs[buf][row * GL_STRIDE + 16 + klo]);
            bfrag[nt] = cat16(lo, hi);
        }
        #pragma unroll
        for (int mt = 0; mt < 2; ++mt)
            #pragma unroll
            for (int nt = 0; nt < 4; ++nt)
                acc[mt][nt] = __builtin_amdgcn_wmma_f32_16x16x32_bf16(
                    false, afrag[mt], false, bfrag[nt],
                    (short)0, acc[mt][nt], false, false);
    }

    // Epilogue: C layout — VGPR r: lanes0-15 M=r, lanes16-31 M=r+8; N = l16
    #pragma unroll
    for (int mt = 0; mt < 2; ++mt)
        #pragma unroll
        for (int nt = 0; nt < 4; ++nt) {
            int n     = n0 + wn + nt * 16 + l16;
            int mbase = m0 + wm + mt * 16 + ((lane < 16) ? 0 : 8);
            #pragma unroll
            for (int r = 0; r < 8; ++r)
                Y[(size_t)(mbase + r) * N + n] = (OutT)acc[mt][nt][r];
        }
}

// ---------------------------------------------------------------------------
// Flash attention (no mask): Q,K,V bf16 [B*T, C] with head h at cols h*128.
// Block: 128 threads = 4 waves; 64 queries per block (16 per wave);
// KV tiles of 64 keys; online softmax; O written as bf16.
// Q and K staged with async global->LDS; V transposed through registers.
// ---------------------------------------------------------------------------
#define QS_STRIDE 136   // 128 + 8 pad
#define VT_STRIDE 72    // 64 + 8 pad
#define PS_STRIDE 72

__global__ __launch_bounds__(128) void flash_attn(
    const bf16* __restrict__ Qg, const bf16* __restrict__ Kg,
    const bf16* __restrict__ Vg, bf16* __restrict__ Og,
    int T, int C, int H, float scale)
{
    __shared__ bf16 Qs[64 * QS_STRIDE];
    __shared__ bf16 Ks[64 * QS_STRIDE];
    __shared__ bf16 Vt[128 * VT_STRIDE];      // V transposed: [d][key]
    __shared__ bf16 Ps[4 * 16 * PS_STRIDE];   // wave-private P regions

    const int tid  = threadIdx.x;
    const int lane = tid & 31;
    const int wave = tid >> 5;
    const int l16  = lane & 15;
    const int klo  = (lane < 16) ? 0 : 8;

    const int qb0  = blockIdx.x * 64;
    const int bh   = blockIdx.y;
    const int b    = bh / H;
    const int h    = bh % H;
    const size_t rowbase = (size_t)b * T;
    const int col0 = h * 128;

    // ---- Stage Q block (64 x 128 bf16) via async copies ----
    #pragma unroll
    for (int c = 0; c < 8; ++c) {
        int chunk = tid + c * 128;       // 0..1023, 16 chunks per row
        int row   = chunk >> 4;
        int part  = (chunk & 15) * 8;
        copy16_g2l(Qg + (rowbase + qb0 + row) * C + col0 + part,
                   &Qs[row * QS_STRIDE + part]);
    }
    async_wait0();
    __syncthreads();

    // ---- Q fragments: 16 rows of this wave, K=128 -> 4 k-steps ----
    v16bf aq[4];
    {
        int row = wave * 16 + l16;
        #pragma unroll
        for (int ks = 0; ks < 4; ++ks) {
            v8bf lo = *(const v8bf*)(&Qs[row * QS_STRIDE + ks * 32 + klo]);
            v8bf hi = *(const v8bf*)(&Qs[row * QS_STRIDE + ks * 32 + 16 + klo]);
            aq[ks] = cat16(lo, hi);
        }
    }

    float mrun[8], lrun[8];
    v8f o[8];
    #pragma unroll
    for (int r = 0; r < 8; ++r) { mrun[r] = -1e30f; lrun[r] = 0.0f; }
    #pragma unroll
    for (int nf = 0; nf < 8; ++nf)
        #pragma unroll
        for (int r = 0; r < 8; ++r) o[nf][r] = 0.0f;

    bf16* Pw = &Ps[wave * 16 * PS_STRIDE];

    const int nkv = T / 64;
    for (int j = 0; j < nkv; ++j) {
        const int kr0 = j * 64;
        __syncthreads();   // previous iteration's reads of Ks/Vt complete

        // ---- Stage K (async, row-major) and V (register path, transposed) ----
        #pragma unroll
        for (int c = 0; c < 8; ++c) {
            int chunk = tid + c * 128;
            int row   = chunk >> 4;
            int part  = (chunk & 15) * 8;
            copy16_g2l(Kg + (rowbase + kr0 + row) * C + col0 + part,
                       &Ks[row * QS_STRIDE + part]);
            v8bf v = *(const v8bf*)(Vg + (rowbase + kr0 + row) * C + col0 + part);
            #pragma unroll
            for (int u = 0; u < 8; ++u)
                Vt[(part + u) * VT_STRIDE + row] = v[u];
        }
        async_wait0();
        __syncthreads();

        // ---- S = Q @ K^T  (16 queries x 64 keys per wave) ----
        v8f s[4];
        #pragma unroll
        for (int nt = 0; nt < 4; ++nt) {
            #pragma unroll
            for (int r = 0; r < 8; ++r) s[nt][r] = 0.0f;
            int row = nt * 16 + l16;     // key index within tile
            #pragma unroll
            for (int ks = 0; ks < 4; ++ks) {
                v8bf lo = *(const v8bf*)(&Ks[row * QS_STRIDE + ks * 32 + klo]);
                v8bf hi = *(const v8bf*)(&Ks[row * QS_STRIDE + ks * 32 + 16 + klo]);
                s[nt] = __builtin_amdgcn_wmma_f32_16x16x32_bf16(
                    false, aq[ks], false, cat16(lo, hi),
                    (short)0, s[nt], false, false);
            }
        }
        #pragma unroll
        for (int nt = 0; nt < 4; ++nt)
            #pragma unroll
            for (int r = 0; r < 8; ++r) s[nt][r] *= scale;

        // ---- Online softmax update ----
        float mnew[8], corr[8], psum[8];
        #pragma unroll
        for (int r = 0; r < 8; ++r) {
            float mx = s[0][r];
            mx = fmaxf(mx, s[1][r]);
            mx = fmaxf(mx, s[2][r]);
            mx = fmaxf(mx, s[3][r]);
            mx = fmaxf(mx, __shfl_xor(mx, 1, 32));
            mx = fmaxf(mx, __shfl_xor(mx, 2, 32));
            mx = fmaxf(mx, __shfl_xor(mx, 4, 32));
            mx = fmaxf(mx, __shfl_xor(mx, 8, 32));
            mnew[r] = fmaxf(mrun[r], mx);
            corr[r] = __expf(mrun[r] - mnew[r]);
            mrun[r] = mnew[r];
            psum[r] = 0.0f;
        }
        #pragma unroll
        for (int nt = 0; nt < 4; ++nt)
            #pragma unroll
            for (int r = 0; r < 8; ++r) {
                float p = __expf(s[nt][r] - mnew[r]);
                s[nt][r] = p;
                psum[r] += p;
            }
        #pragma unroll
        for (int r = 0; r < 8; ++r) {
            float ps = psum[r];
            ps += __shfl_xor(ps, 1, 32);
            ps += __shfl_xor(ps, 2, 32);
            ps += __shfl_xor(ps, 4, 32);
            ps += __shfl_xor(ps, 8, 32);
            lrun[r] = lrun[r] * corr[r] + ps;
        }
        #pragma unroll
        for (int nf = 0; nf < 8; ++nf)
            #pragma unroll
            for (int r = 0; r < 8; ++r) o[nf][r] *= corr[r];

        // ---- P: C-layout -> A-layout via wave-private LDS ----
        #pragma unroll
        for (int nt = 0; nt < 4; ++nt)
            #pragma unroll
            for (int r = 0; r < 8; ++r) {
                int prow = r + ((lane < 16) ? 0 : 8);
                Pw[prow * PS_STRIDE + nt * 16 + l16] = (bf16)s[nt][r];
            }
        v16bf ap[2];
        #pragma unroll
        for (int ks = 0; ks < 2; ++ks) {
            v8bf lo = *(const v8bf*)(&Pw[l16 * PS_STRIDE + ks * 32 + klo]);
            v8bf hi = *(const v8bf*)(&Pw[l16 * PS_STRIDE + ks * 32 + 16 + klo]);
            ap[ks] = cat16(lo, hi);
        }

        // ---- O += P @ V  (K = 64 keys, N = 128 dims) ----
        #pragma unroll
        for (int nf = 0; nf < 8; ++nf) {
            int d = nf * 16 + l16;
            #pragma unroll
            for (int ks = 0; ks < 2; ++ks) {
                v8bf lo = *(const v8bf*)(&Vt[d * VT_STRIDE + ks * 32 + klo]);
                v8bf hi = *(const v8bf*)(&Vt[d * VT_STRIDE + ks * 32 + 16 + klo]);
                o[nf] = __builtin_amdgcn_wmma_f32_16x16x32_bf16(
                    false, ap[ks], false, cat16(lo, hi),
                    (short)0, o[nf], false, false);
            }
        }
    }

    // ---- Epilogue: O /= l, store bf16 at [b*T + q, h*128 + d] ----
    #pragma unroll
    for (int nf = 0; nf < 8; ++nf) {
        int n = col0 + nf * 16 + l16;
        #pragma unroll
        for (int r = 0; r < 8; ++r) {
            size_t m = rowbase + qb0 + wave * 16 + r + ((lane < 16) ? 0 : 8);
            Og[m * C + n] = (bf16)(o[nf][r] / lrun[r]);
        }
    }
}

// ---------------------------------------------------------------------------
// Host launcher
// ---------------------------------------------------------------------------
extern "C" void kernel_launch(void* const* d_in, const int* in_sizes, int n_in,
                              void* d_out, int out_size, void* d_ws, size_t ws_size,
                              hipStream_t stream) {
    const int B = 2, T = 2048, C = 2048, H = 16;
    const int M = B * T;                 // 4096
    const float scale = 0.088388347648318447f;  // 1/sqrt(128)

    const float* x  = (const float*)d_in[0];
    const float* Wq = (const float*)d_in[1];
    const float* Wk = (const float*)d_in[2];
    const float* Wv = (const float*)d_in[3];
    const float* Wo = (const float*)d_in[4];

    char* ws = (char*)d_ws;
    size_t off = 0;
    bf16* xb  = (bf16*)(ws + off); off += (size_t)M * C * sizeof(bf16);
    bf16* wqb = (bf16*)(ws + off); off += (size_t)C * C * sizeof(bf16);
    bf16* wkb = (bf16*)(ws + off); off += (size_t)C * C * sizeof(bf16);
    bf16* wvb = (bf16*)(ws + off); off += (size_t)C * C * sizeof(bf16);
    bf16* wob = (bf16*)(ws + off); off += (size_t)C * C * sizeof(bf16);
    bf16* qb  = (bf16*)(ws + off); off += (size_t)M * C * sizeof(bf16);
    bf16* kb  = (bf16*)(ws + off); off += (size_t)M * C * sizeof(bf16);
    bf16* vb  = (bf16*)(ws + off); off += (size_t)M * C * sizeof(bf16);
    bf16* ob  = (bf16*)(ws + off); off += (size_t)M * C * sizeof(bf16);

    // 1) f32 -> bf16 conversions
    {
        int n4 = (M * C) / 4;
        cvt_f32_to_bf16_x4<<<(n4 + 255) / 256, 256, 0, stream>>>((const v4f*)x, (v4bf*)xb, n4);
        int w4 = (C * C) / 4;
        cvt_f32_to_bf16_x4<<<(w4 + 255) / 256, 256, 0, stream>>>((const v4f*)Wq, (v4bf*)wqb, w4);
        cvt_f32_to_bf16_x4<<<(w4 + 255) / 256, 256, 0, stream>>>((const v4f*)Wk, (v4bf*)wkb, w4);
        cvt_f32_to_bf16_x4<<<(w4 + 255) / 256, 256, 0, stream>>>((const v4f*)Wv, (v4bf*)wvb, w4);
        cvt_f32_to_bf16_x4<<<(w4 + 255) / 256, 256, 0, stream>>>((const v4f*)Wo, (v4bf*)wob, w4);
    }

    // 2) Q/K/V projections: [4096,2048] = xb @ W^T
    dim3 ggrid(C / 128, M / 128);
    gemm_xwT<bf16><<<ggrid, 256, 0, stream>>>(xb, wqb, qb, M, C, C);
    gemm_xwT<bf16><<<ggrid, 256, 0, stream>>>(xb, wkb, kb, M, C, C);
    gemm_xwT<bf16><<<ggrid, 256, 0, stream>>>(xb, wvb, vb, M, C, C);

    // 3) Flash attention: grid = (query blocks, B*H)
    flash_attn<<<dim3(T / 64, B * H), 128, 0, stream>>>(qb, kb, vb, ob, T, C, H, scale);

    // 4) Output projection -> f32 d_out
    gemm_xwT<float><<<ggrid, 256, 0, stream>>>(ob, wob, (float*)d_out, M, C, C);
}
